// DistancePredictor_58076547777195
// MI455X (gfx1250) — matrix operand order repaired and checked by reading, not compiled, not run on hardware
//
#include <hip/hip_runtime.h>
#include <hip/hip_bf16.h>
#include <math.h>

#define BB 2
#define LL 1024
#define DD 128

typedef __attribute__((ext_vector_type(16))) _Float16 v16h;
typedef __attribute__((ext_vector_type(8)))  _Float16 v8h;
typedef __attribute__((ext_vector_type(8)))  float    v8f;

union U16h { v16h v; v8h h[2]; };

// Branch-free exact-GELU: 0.5*x*(1+erf(x/sqrt(2))) with erf via
// Abramowitz-Stegun 7.1.26 (|err| <= 1.5e-7, well under the f16 WMMA noise).
// Cost: 1 v_rcp_f32 + 1 v_exp_f32 + ~9 FMA, no exec divergence.
__device__ __forceinline__ float gelu_exact(float x) {
    const float u  = x * 0.70710678118654752f;  // x / sqrt(2)
    const float ax = __builtin_fabsf(u);
    const float t  = __builtin_amdgcn_rcpf(fmaf(0.3275911f, ax, 1.0f));
    float p = fmaf(1.061405429f, t, -1.453152027f);
    p = fmaf(p, t, 1.421413741f);
    p = fmaf(p, t, -0.284496736f);
    p = fmaf(p, t, 0.254829592f);
    const float e = __expf(-u * u);
    const float y = fmaf(-p * t, e, 1.0f);      // erf(|u|)
    const float erfv = __builtin_copysignf(y, u);
    return 0.5f * x * (1.0f + erfv);
}

// ---------------------------------------------------------------------------
// Kernel 1: per-row prep.
//   Abuf[n,h] = sum_d X[n,d]*(Wi[d,h]+Wd[d,h]) + b1[h]
//   Bbuf[n,h] = sum_d X[n,d]*(Wj[d,h]-Wd[d,h])
//   Xh[n,h]   = (f16) X[n,h]
// ---------------------------------------------------------------------------
__global__ void __launch_bounds__(128) dp_prep_kernel(
    const float* __restrict__ X, const float* __restrict__ W1,
    const float* __restrict__ b1,
    float* __restrict__ Abuf, float* __restrict__ Bbuf,
    _Float16* __restrict__ Xh)
{
    __shared__ float xs[DD];
    const int n = blockIdx.x;
    const int h = threadIdx.x;
    xs[h] = X[(size_t)n * DD + h];
    __syncthreads();

    float accA = b1[h];
    float accB = 0.0f;
    #pragma unroll 4
    for (int d = 0; d < DD; ++d) {
        const float xd = xs[d];
        const float wi = W1[(0 * DD + d) * DD + h];
        const float wj = W1[(1 * DD + d) * DD + h];
        const float wd = W1[(2 * DD + d) * DD + h];
        accA = fmaf(xd, wi + wd, accA);
        accB = fmaf(xd, wj - wd, accB);
    }
    Abuf[(size_t)n * DD + h] = accA;
    Bbuf[(size_t)n * DD + h] = accB;
    Xh[(size_t)n * DD + h] = (_Float16)xs[h];
}

// ---------------------------------------------------------------------------
// Kernel 2: main pairwise WMMA kernel. One workgroup per (b,i) row.
//   M_i[d,h] = X[i,d] * Wp[d,h]   (built f16, h-major in LDS: MiT[h][d])
//   For each 16-wide j-tile:  D(16j x 16h) += Xh[jtile] @ M_i  via
//   v_wmma_f32_16x16x32_f16 over 4 K-chunks, then GELU + *W2[h] folded
//   into per-(j) scalars, cross-lane reduced, relu(+b2) stored to out.
// ---------------------------------------------------------------------------
__global__ void __launch_bounds__(256) dp_main_kernel(
    const float* __restrict__ X, const float* __restrict__ W1,
    const float* __restrict__ W2, const float* __restrict__ b2,
    const float* __restrict__ Abuf, const float* __restrict__ Bbuf,
    const _Float16* __restrict__ Xh, float* __restrict__ out)
{
    __shared__ __align__(16) _Float16 MiT[DD * DD]; // MiT[h*128 + d], 32 KB
    __shared__ float xs[DD];
    __shared__ float Ald[DD];
    __shared__ float w2ld[DD];

    const int n = blockIdx.x;          // n = b*L + i
    const int b = n >> 10;
    const int tid = threadIdx.x;

    if (tid < DD) {
        xs[tid]   = X[(size_t)n * DD + tid];
        Ald[tid]  = Abuf[(size_t)n * DD + tid]; // b1 already folded in
        w2ld[tid] = W2[tid];
    }
    __syncthreads();

    // Build MiT[h][d] = X[i,d] * Wp[d,h];  Wp = W1 rows [384,512)
    for (int k = tid; k < DD * DD; k += 256) {
        const int h = k >> 7;
        const int d = k & 127;
        MiT[k] = (_Float16)(xs[d] * W1[(3 * DD + d) * DD + h]);
    }
    __syncthreads();

    const int wave = tid >> 5;
    const int lane = tid & 31;
    const int nn = lane & 15;   // D column (h) / A row (j) within tile
    const int s  = lane >> 4;   // half-wave select
    const float b2v = b2[0];

    // 64 j-tiles of 16; 8 waves -> 8 tiles per wave
    for (int jt = wave; jt < 64; jt += 8) {
        const int jb = jt * 16;

        // A fragments: Xh row (j = jb + nn), four K-chunks of 32.
        // halfs 0..7  <- K = 32*kc + 8*s + (0..7)
        // halfs 8..15 <- K = 32*kc + 16 + 8*s + (0..7)
        const v8h* xp = (const v8h*)(Xh + ((size_t)(b * LL + jb + nn)) * DD);
        v16h af[4];
        #pragma unroll
        for (int kc = 0; kc < 4; ++kc) {
            U16h a;
            a.h[0] = xp[4 * kc + s];
            a.h[1] = xp[4 * kc + s + 2];
            af[kc] = a.v;
        }

        float outacc[8];
        #pragma unroll
        for (int r = 0; r < 8; ++r) outacc[r] = 0.0f;

        #pragma unroll 1
        for (int ht = 0; ht < 8; ++ht) {
            v8f c = {};
            #pragma unroll
            for (int kc = 0; kc < 4; ++kc) {
                // B fragment: column = nn, halfs 0..7 <- K = 32*kc + 8*s + t, halfs 8..15 <- K = 32*kc + 16 + 8*s + t
                U16h bu;
                const v8h* bp =
                    (const v8h*)&MiT[(ht * 16 + nn) * DD + kc * 32 + s * 8];
                bu.h[0] = bp[0];
                bu.h[1] = bp[2];
                c = __builtin_amdgcn_wmma_f32_16x16x32_f16(
                        false, af[kc], false, bu.v, (short)0, c, false, false);
            }
            const int h = ht * 16 + nn;
            const float aval = Ald[h];
            const float w2v  = w2ld[h];
            #pragma unroll
            for (int r = 0; r < 8; ++r) {
                const int j = jb + r + 8 * s;               // D row M
                const float val = c[r] + aval
                          + Bbuf[((size_t)(b * LL + j)) * DD + h];
                outacc[r] = fmaf(gelu_exact(val), w2v, outacc[r]);
            }
        }

        // sum over the 16 h-lanes of each half-wave (j stays in-group)
        #pragma unroll
        for (int r = 0; r < 8; ++r) {
            float v = outacc[r];
            v += __shfl_xor(v, 1, 32);
            v += __shfl_xor(v, 2, 32);
            v += __shfl_xor(v, 4, 32);
            v += __shfl_xor(v, 8, 32);
            outacc[r] = v;
        }

        if (nn < 8) {
            const int j = jb + nn + 8 * s;
            float v = 0.0f;
            #pragma unroll
            for (int r = 0; r < 8; ++r)
                if (r == nn) v = outacc[r];
            out[(size_t)n * LL + j] = fmaxf(v + b2v, 0.0f);
        }
    }
}

// ---------------------------------------------------------------------------
// Kernel 3: symmetrize in place and zero the diagonal.
// Thread owning (b,i,j) with i<j reads/writes both mirror elements (race-free).
// ---------------------------------------------------------------------------
__global__ void __launch_bounds__(256) dp_sym_kernel(float* __restrict__ out)
{
    const int idx = blockIdx.x * 256 + threadIdx.x;
    const int total = BB * LL * LL;
    if (idx >= total) return;
    const int b = idx >> 20;
    const int rem = idx & (LL * LL - 1);
    const int i = rem >> 10;
    const int j = rem & (LL - 1);
    float* base = out + ((size_t)b * LL * LL);
    if (i == j) {
        base[(size_t)i * LL + j] = 0.0f;
    } else if (i < j) {
        const float a = base[(size_t)i * LL + j];
        const float c = base[(size_t)j * LL + i];
        const float m = 0.5f * (a + c);
        base[(size_t)i * LL + j] = m;
        base[(size_t)j * LL + i] = m;
    }
}

extern "C" void kernel_launch(void* const* d_in, const int* in_sizes, int n_in,
                              void* d_out, int out_size, void* d_ws, size_t ws_size,
                              hipStream_t stream) {
    const float* X  = (const float*)d_in[0]; // (B,L,128)
    const float* W1 = (const float*)d_in[1]; // (512,128)
    const float* b1 = (const float*)d_in[2]; // (128,)
    const float* W2 = (const float*)d_in[3]; // (128,1)
    const float* b2 = (const float*)d_in[4]; // (1,)
    float* out = (float*)d_out;              // (B,L,L)

    char* ws = (char*)d_ws;
    float*    Abuf = (float*)ws;                       // 1 MB
    float*    Bbuf = (float*)(ws + (1u << 20));        // 1 MB
    _Float16* Xh   = (_Float16*)(ws + (2u << 20));     // 0.5 MB

    dp_prep_kernel<<<BB * LL, 128, 0, stream>>>(X, W1, b1, Abuf, Bbuf, Xh);
    dp_main_kernel<<<BB * LL, 256, 0, stream>>>(X, W1, W2, b2, Abuf, Bbuf, Xh, out);
    dp_sym_kernel<<<(BB * LL * LL + 255) / 256, 256, 0, stream>>>(out);
}
